// ECTLoss_19232863551765
// MI455X (gfx1250) — matrix operand order repaired and checked by reading, not compiled, not run on hardware
//
#include <hip/hip_runtime.h>

typedef __attribute__((ext_vector_type(16))) _Float16 v16h;
typedef __attribute__((ext_vector_type(8)))  float    v8f;
typedef __attribute__((ext_vector_type(2)))  _Float16 h2;

#define B_      2
#define C_      3
#define S_      48
#define P_TOT   (S_*S_*S_)        // 110592
#define D_      32
#define R_      64
#define NCOL    (R_*D_)           // 2048
#define NROW    (B_*C_)           // 6 valid rows, padded to 16
#define NSLICE  32
#define PS      (P_TOT/NSLICE)    // 3456
#define NCHUNK  (PS/32)           // 108
#define NTILE   (NCOL/16)         // 128
#define NWAVES  (NTILE*NSLICE)    // 4096

// workspace layout (bytes)
constexpr size_t OFF_DIRS = 0;                                   // 96 f32
constexpr size_t OFF_ACC  = 4096;                                // 12288 f32
constexpr size_t OFF_WA   = 65536;                               // 16*P f16 = 3538944 B
constexpr size_t OFF_NH   = OFF_WA + (size_t)16*P_TOT*2 + 65536; // 32*P f32 = 14155776 B

// ---------------- direction generation: JAX threefry2x32 + erfinv ----------
__device__ __forceinline__ unsigned rotl32(unsigned x, int r) {
    return (x << r) | (x >> (32 - r));
}

__device__ float erfinv_f32(float x) {  // XLA ErfInv32 polynomial (Giles)
    float w = -logf((1.0f - x) * (1.0f + x));
    float p;
    if (w < 5.0f) {
        w = w - 2.5f;
        p =               2.81022636e-08f;
        p = fmaf(p, w,    3.43273939e-07f);
        p = fmaf(p, w,   -3.5233877e-06f);
        p = fmaf(p, w,   -4.39150654e-06f);
        p = fmaf(p, w,    0.00021858087f);
        p = fmaf(p, w,   -0.00125372503f);
        p = fmaf(p, w,   -0.00417768164f);
        p = fmaf(p, w,    0.246640727f);
        p = fmaf(p, w,    1.50140941f);
    } else {
        w = sqrtf(w) - 3.0f;
        p =              -0.000200214257f;
        p = fmaf(p, w,    0.000100950558f);
        p = fmaf(p, w,    0.00134934322f);
        p = fmaf(p, w,   -0.00367342844f);
        p = fmaf(p, w,    0.00573950773f);
        p = fmaf(p, w,   -0.0076224613f);
        p = fmaf(p, w,    0.00943887047f);
        p = fmaf(p, w,    1.00167406f);
        p = fmaf(p, w,    2.83297682f);
    }
    return p * x;
}

__global__ void k_dirs(float* __restrict__ dirs) {  // dirs[3][32]
    __shared__ float raw[96];
    __shared__ float nrm[32];
    int i = threadIdx.x;
    if (i < 96) {
        // counts 0..95 split into x0=counts[0:48], x1=counts[48:96]
        unsigned pos = (i < 48) ? (unsigned)i : (unsigned)(i - 48);
        unsigned x0 = pos, x1 = pos + 48u;
        const unsigned ks0 = 0u, ks1 = 17u, ks2 = 0x1BD11BDAu ^ 0u ^ 17u;
        x0 += ks0; x1 += ks1;
#define TF4(a,b,c,d) \
        x0 += x1; x1 = rotl32(x1,(a)); x1 ^= x0; \
        x0 += x1; x1 = rotl32(x1,(b)); x1 ^= x0; \
        x0 += x1; x1 = rotl32(x1,(c)); x1 ^= x0; \
        x0 += x1; x1 = rotl32(x1,(d)); x1 ^= x0;
        TF4(13,15,26,6);  x0 += ks1; x1 += ks2 + 1u;
        TF4(17,29,16,24); x0 += ks2; x1 += ks0 + 2u;
        TF4(13,15,26,6);  x0 += ks0; x1 += ks1 + 3u;
        TF4(17,29,16,24); x0 += ks1; x1 += ks2 + 4u;
        TF4(13,15,26,6);  x0 += ks2; x1 += ks0 + 5u;
#undef TF4
        unsigned bits = (i < 48) ? x0 : x1;
        // uniform(lo=nextafter(-1,0), hi=1): bits>>9 | 1.0f, -1 -> [0,1)
        unsigned fb = (bits >> 9) | 0x3f800000u;
        float f = __uint_as_float(fb) - 1.0f;
        const float LO = -0.99999994f;            // nextafterf(-1,0)
        float span = 1.0f - LO;                   // rounds to 2.0f in f32
        float u = f * span + LO;
        u = fmaxf(LO, u);
        raw[i] = erfinv_f32(u) * 1.41421356f;     // sqrt(2) in f32
    }
    __syncthreads();
    if (i < 32) {
        float s = raw[i]*raw[i] + raw[32+i]*raw[32+i] + raw[64+i]*raw[64+i];
        nrm[i] = fmaxf(sqrtf(s), 1e-12f);
    }
    __syncthreads();
    if (i < 96) dirs[i] = raw[i] / nrm[i & 31];
}

// ------ nhs[d][p] = -4 * (coords[p] . dirs[:,d])  (pre-scaled half-arg) ----
__global__ void k_nh(const float* __restrict__ dirs, float* __restrict__ nhs) {
    int idx = blockIdx.x * blockDim.x + threadIdx.x;
    if (idx >= D_ * P_TOT) return;
    int d = idx / P_TOT;
    int p = idx - d * P_TOT;
    int z = p % S_;
    int y = (p / S_) % S_;
    int x = p / (S_ * S_);
    const float st = 2.0f / 47.0f;
    float cx = -1.0f + st * (float)x;
    float cy = -1.0f + st * (float)y;
    float cz = -1.0f + st * (float)z;
    float nh = cx * dirs[d] + cy * dirs[32 + d] + cz * dirs[64 + d];
    nhs[(size_t)d * P_TOT + p] = -4.0f * nh;   // half of SCALE=8
}

// ------- w_diff rows (softmax - onehot) as f16, pad rows 6..15, zero acc ---
__global__ void k_wdiff(const float* __restrict__ pred, const int* __restrict__ tgt,
                        _Float16* __restrict__ wA, float* __restrict__ acc) {
    int idx = blockIdx.x * blockDim.x + threadIdx.x;
    if (idx < NROW * NCOL) acc[idx] = 0.0f;       // zero the 6x2048 accumulator
    if (idx >= B_ * P_TOT) return;
    int b = idx / P_TOT;
    int p = idx - b * P_TOT;
    size_t base = (size_t)b * C_ * P_TOT + p;
    float a0 = pred[base];
    float a1 = pred[base + P_TOT];
    float a2 = pred[base + 2 * P_TOT];
    float mx = fmaxf(a0, fmaxf(a1, a2));
    float e0 = __expf(a0 - mx), e1 = __expf(a1 - mx), e2 = __expf(a2 - mx);
    float inv = 1.0f / (e0 + e1 + e2);
    int t = tgt[idx];
    size_t row = (size_t)b * C_ * P_TOT + p;      // rows m=b*3+c, stride P
    wA[row]             = (_Float16)(e0 * inv - (t == 0 ? 1.0f : 0.0f));
    wA[row + P_TOT]     = (_Float16)(e1 * inv - (t == 1 ? 1.0f : 0.0f));
    wA[row + 2 * P_TOT] = (_Float16)(e2 * inv - (t == 2 ? 1.0f : 0.0f));
    if (b == 0) {                                  // zero padded rows 6..15
        for (int m = NROW; m < 16; ++m) wA[(size_t)m * P_TOT + p] = (_Float16)0.0f;
    }
}

// sigmoid(2h) = 0.5*tanh(h) + 0.5  (gfx1250 v_tanh_f32; matches XLA logistic)
__device__ __forceinline__ float sigmoid_from_half_arg(float h) {
#if __has_builtin(__builtin_amdgcn_tanhf)
    return fmaf(0.5f, __builtin_amdgcn_tanhf(h), 0.5f);
#elif __has_builtin(__builtin_amdgcn_tanh_f32)
    return fmaf(0.5f, __builtin_amdgcn_tanh_f32(h), 0.5f);
#else
    float e = __expf(-2.0f * h);
    return __builtin_amdgcn_rcpf(1.0f + e);       // v_rcp_f32, no IEEE div chain
#endif
}

// ---------------- WMMA GEMM: acc[m][r*32+d] += sum_p wA[m][p]*ecc ----------
__global__ void k_gemm(const _Float16* __restrict__ wA,
                       const float* __restrict__ nhs,
                       float* __restrict__ acc) {
    int wave  = (int)((blockIdx.x * blockDim.x + threadIdx.x) >> 5); // 0..NWAVES-1
    int lane  = threadIdx.x & 31;
    int tile  = wave / NSLICE;                // 0..127  -> 16 output columns
    int slice = wave - tile * NSLICE;         // P-slice
    int r     = tile >> 1;                    // one threshold per tile
    int d0    = (tile & 1) << 4;              // 16 consecutive directions
    int half  = lane >> 4;
    int ln    = lane & 15;
    int d     = d0 + ln;                      // this lane's direction (B/N role)

    const float RADIUS = 1.1f * 1.73205081f;  // 1.1*sqrt(3)
    float linr = -RADIUS + (2.0f * RADIUS / 63.0f) * (float)r;
    float s4   = 4.0f * linr;                 // half of SCALE * lin[r]
    int p0base = slice * PS;
    const float* nhrow = nhs + (size_t)d * P_TOT;

    v8f c = {};
    for (int ch = 0; ch < NCHUNK; ++ch) {
        int p0 = p0base + ch * 32;
        v16h a, b;
#pragma unroll
        for (int v = 0; v < 8; ++v) {
            // A fragment: 16-bit A 16x32 layout (ISA 7.12.2): row m=lane&15,
            // K pair = 16*(v>>2) + 2*(v&3) + 8*half  (contiguous -> b128 loads)
            int kA = ((v >> 2) << 4) + ((v & 3) << 1) + (half << 3);
            h2 av = *(const h2*)(wA + (size_t)ln * P_TOT + p0 + kA);
            a[2 * v]     = av.x;
            a[2 * v + 1] = av.y;
            // B fragment: 32x16, same K order as A (K pair = 16*(v>>2) + 2*(v&3) + 8*half)
            int kB = ((v >> 2) << 4) + ((v & 3) << 1) + (half << 3);
            float h0 = s4 + nhrow[p0 + kB];       // 4*(lin[r] - nh)
            float h1 = s4 + nhrow[p0 + kB + 1];
            b[2 * v]     = (_Float16)sigmoid_from_half_arg(h0);
            b[2 * v + 1] = (_Float16)sigmoid_from_half_arg(h1);
        }
        c = __builtin_amdgcn_wmma_f32_16x16x32_f16(
                /*neg_a=*/false, a, /*neg_b=*/false, b,
                /*c_mod=*/(short)0, c, /*reuse_a=*/false, /*reuse_b=*/false);
    }
    // D layout: VGPR v holds M = v + 8*half, N = lane&15
    int col = tile * 16 + ln;
#pragma unroll
    for (int v = 0; v < 8; ++v) {
        int m = v + (half << 3);
        if (m < NROW) atomicAdd(acc + (size_t)m * NCOL + col, c[v]);
    }
}

// ---------------- loss = mean(acc^2) over 12288 elements -------------------
__global__ void k_final(const float* __restrict__ acc, float* __restrict__ out) {
    __shared__ float red[256];
    int t = threadIdx.x;
    float s = 0.0f;
    for (int i = t; i < NROW * NCOL; i += 256) {
        float v = acc[i];
        s += v * v;
    }
    red[t] = s;
    __syncthreads();
    for (int w = 128; w > 0; w >>= 1) {
        if (t < w) red[t] += red[t + w];
        __syncthreads();
    }
    if (t == 0) out[0] = red[0] / (float)(NROW * NCOL);
}

extern "C" void kernel_launch(void* const* d_in, const int* in_sizes, int n_in,
                              void* d_out, int out_size, void* d_ws, size_t ws_size,
                              hipStream_t stream) {
    (void)in_sizes; (void)n_in; (void)out_size; (void)ws_size;
    const float* pred = (const float*)d_in[0];
    const int*   tgt  = (const int*)d_in[1];
    float*       out  = (float*)d_out;

    char* ws = (char*)d_ws;
    float*    dirs = (float*)(ws + OFF_DIRS);
    float*    acc  = (float*)(ws + OFF_ACC);
    _Float16* wA   = (_Float16*)(ws + OFF_WA);
    float*    nhs  = (float*)(ws + OFF_NH);

    k_dirs<<<1, 128, 0, stream>>>(dirs);

    int nNH = D_ * P_TOT;
    k_nh<<<(nNH + 255) / 256, 256, 0, stream>>>(dirs, nhs);

    int nW = B_ * P_TOT;
    k_wdiff<<<(nW + 255) / 256, 256, 0, stream>>>(pred, tgt, wA, acc);

    // 4096 waves, 8 waves (256 threads) per block
    k_gemm<<<NWAVES / 8, 256, 0, stream>>>(wA, nhs, acc);

    k_final<<<1, 256, 0, stream>>>(acc, out);
}
